// DGCNN_cls_14877766713519
// MI455X (gfx1250) — matrix operand order repaired and checked by reading, not compiled, not run on hardware
//
#include <hip/hip_runtime.h>
#include <hip/hip_bf16.h>

typedef __attribute__((ext_vector_type(16))) _Float16 v16h;
typedef __attribute__((ext_vector_type(8)))  float    v8f;

#define EPS_BN 1e-5f
#define SLOPE 0.2f

// ---- ordered-uint encoding for float max via atomicMax(unsigned) ----
__device__ __forceinline__ unsigned fkey(float f) {
    unsigned u = __float_as_uint(f);
    return (u & 0x80000000u) ? ~u : (u | 0x80000000u);
}
__device__ __forceinline__ float fkeyinv(unsigned u) {
    return (u & 0x80000000u) ? __uint_as_float(u ^ 0x80000000u)
                             : __uint_as_float(~u);
}

// ---------------------------------------------------------------------------
// Edge-conv GEMM: per block -> one b, 16 points (NB) x 20 neighbors = 320
// positions. Gathers [nbr-ctr; ctr] features to LDS (f16), runs WMMA GEMM
// W(COUT x KPAD) x feat(KPAD x 320). Each wave is pinned to one channel tile
// (mt): A fragments live in VGPRs for the whole position loop, per-channel
// sum/sumsq accumulate in registers, max-over-K (pre-BN, monotone) via LDS
// ordered-uint atomicMax.
// ---------------------------------------------------------------------------
template<int CIN, int COUT, int KPAD>
__launch_bounds__(256)
__global__ void edgeconv_wmma(const float* __restrict__ xprev,
                              const long long* __restrict__ idx,
                              const _Float16* __restrict__ Wh,
                              float* __restrict__ ymax,
                              float* __restrict__ gsum,
                              float* __restrict__ gsumsq,
                              int Nn) {
    constexpr int NB   = 16;           // points per block
    constexpr int KNEI = 20;           // neighbors
    constexpr int POS  = NB * KNEI;    // 320 positions
    constexpr int PT   = POS / 16;     // 20 position tiles
    constexpr int MT   = COUT / 16;    // channel tiles
    constexpr int KB   = KPAD / 32;    // K blocks per WMMA chain
    constexpr int MPW  = (MT + 7) / 8;             // mt iterations per wave
    constexpr int WPM  = (MT >= 8) ? 1 : (8 / MT); // waves sharing one mt

    extern __shared__ char smem[];
    _Float16* sfeat = (_Float16*)smem;                                  // POS*KPAD
    unsigned* smax  = (unsigned*)(smem + (size_t)POS * KPAD * sizeof(_Float16));
    float*    ssum  = (float*)(smax + COUT * NB);                       // COUT
    float*    ssq   = ssum + COUT;                                      // COUT
    float*    sctr  = ssq + COUT;                                       // NB*CIN

    const int tid = threadIdx.x;
    const int b   = blockIdx.x / (Nn / NB);
    const int n0  = (blockIdx.x % (Nn / NB)) * NB;
    const float* xb = xprev + (size_t)b * CIN * Nn;

    for (int i = tid; i < COUT * NB; i += 256) smax[i] = 0u;   // key lower bound
    for (int i = tid; i < 2 * COUT; i += 256) ssum[i] = 0.f;
    // stage the 16 center rows once (each is reused by 20 neighbors)
    for (int i = tid; i < NB * CIN; i += 256) {
        const int ln = i / CIN, c = i % CIN;
        sctr[i] = xb[c * Nn + n0 + ln];
    }
    __syncthreads();

    // ---- gather + feature build into LDS (f16, row = position) ----
    for (int p = tid; p < POS; p += 256) {
        const int ln = p / KNEI, kk = p % KNEI;
        const int n  = n0 + ln;
        const int j  = (int)idx[(size_t)n * KNEI + kk];
        _Float16* f  = sfeat + (size_t)p * KPAD;
#pragma unroll
        for (int c = 0; c < CIN; ++c) {
            const float ctr = sctr[ln * CIN + c];
            const float nbr = xb[c * Nn + j];
            f[c]       = (_Float16)(nbr - ctr);
            f[CIN + c] = (_Float16)ctr;
        }
#pragma unroll
        for (int c = 2 * CIN; c < KPAD; ++c) f[c] = (_Float16)0.f;
    }
    __syncthreads();

    const int wave = tid >> 5, lane = tid & 31;
    const int lrow = lane & 15;     // row (A) / column (B,D) within tile
    const int lhi  = lane >> 4;     // half-wave select

#pragma unroll
    for (int mi = 0; mi < MPW; ++mi) {
        const int mt  = (MT >= 8) ? (wave + mi * 8) : (wave / WPM);
        const int pt0 = (MT >= 8) ? 0 : (wave % WPM);
        const int ptS = (MT >= 8) ? 1 : WPM;

        // ---- preload A fragments (weights) for this mt into registers ----
        // lane<16: K in {e, 16+e | e<8}; lane>=16 shifted by +8 within 32-block.
        v16h afrag[KB];
#pragma unroll
        for (int kb = 0; kb < KB; ++kb) {
            const _Float16* wrow =
                Wh + (size_t)(mt * 16 + lrow) * KPAD + kb * 32 + lhi * 8;
#pragma unroll
            for (int e = 0; e < 16; ++e)
                afrag[kb][e] = wrow[(e & 7) + ((e >> 3) << 4)];
        }

        const int mbase = mt * 16 + lhi * 8;
        float rsum[8], rsq[8];
#pragma unroll
        for (int r = 0; r < 8; ++r) { rsum[r] = 0.f; rsq[r] = 0.f; }

        for (int pt = pt0; pt < PT; pt += ptS) {
            v8f acc = {};
#pragma unroll
            for (int kb = 0; kb < KB; ++kb) {
                // B fragment: feat, column = position; K in {e, 16+e | e<8}, +8 for upper half-wave.
                v16h bf;
                const _Float16* fp =
                    sfeat + (size_t)(pt * 16 + lrow) * KPAD + kb * 32 + lhi * 8;
#pragma unroll
                for (int e = 0; e < 16; ++e) bf[e] = fp[(e & 7) + ((e >> 3) << 4)];
                acc = __builtin_amdgcn_wmma_f32_16x16x32_f16(
                    false, afrag[kb], false, bf, (short)0, acc, false, false);
            }
            // D layout: lane owns column (position) lrow; rows mbase..mbase+7
            const int p  = pt * 16 + lrow;
            const int ln = p / KNEI;
#pragma unroll
            for (int r = 0; r < 8; ++r) {
                const float y = acc[r];
                rsum[r] += y;
                rsq[r]  += y * y;
                atomicMax(&smax[(mbase + r) * NB + ln], fkey(y));
            }
        }
#pragma unroll
        for (int r = 0; r < 8; ++r) {
            atomicAdd(&ssum[mbase + r], rsum[r]);
            atomicAdd(&ssq[mbase + r],  rsq[r]);
        }
    }
    __syncthreads();

    for (int i = tid; i < COUT * NB; i += 256) {
        const int ch = i / NB, ln = i % NB;
        ymax[((size_t)b * COUT + ch) * Nn + n0 + ln] = fkeyinv(smax[i]);
    }
    for (int ch = tid; ch < COUT; ch += 256) {
        atomicAdd(&gsum[ch],   ssum[ch]);
        atomicAdd(&gsumsq[ch], ssq[ch]);
    }
}

// ---------------------------------------------------------------------------
// Layer 4: plain 1x1 conv over concatenated [x1;x2;x3] (256 ch), stores full
// pre-BN y (needed elementwise later) + per-channel stats. Same wave-pinned
// schedule: each wave owns 2 channel tiles, A fragments in registers.
// ---------------------------------------------------------------------------
template<int CTOT, int COUT>
__launch_bounds__(256)
__global__ void conv1d_wmma(const float* __restrict__ x1,
                            const float* __restrict__ x2,
                            const float* __restrict__ x3,
                            const _Float16* __restrict__ Wh,
                            float* __restrict__ ybuf,
                            float* __restrict__ gsum,
                            float* __restrict__ gsumsq,
                            int Nn) {
    constexpr int NBP = 64;            // positions per block
    constexpr int PT  = NBP / 16;      // 4
    constexpr int MT  = COUT / 16;     // 16
    constexpr int KB  = CTOT / 32;     // 8
    constexpr int MPW = MT / 8;        // 2 mt per wave

    extern __shared__ char smem[];
    _Float16* sfeat = (_Float16*)smem;                                   // NBP*CTOT
    float*    ssum  = (float*)(smem + (size_t)NBP * CTOT * sizeof(_Float16));
    float*    ssq   = ssum + COUT;

    const int tid = threadIdx.x;
    const int b   = blockIdx.x / (Nn / NBP);
    const int n0  = (blockIdx.x % (Nn / NBP)) * NBP;

    for (int i = tid; i < 2 * COUT; i += 256) ssum[i] = 0.f;

    for (int i = tid; i < NBP * CTOT; i += 256) {
        const int p = i / CTOT, c = i % CTOT;
        const int n = n0 + p;
        float v;
        if (c < 64)       v = x1[((size_t)b * 64  +  c       ) * Nn + n];
        else if (c < 128) v = x2[((size_t)b * 64  + (c - 64) ) * Nn + n];
        else              v = x3[((size_t)b * 128 + (c - 128)) * Nn + n];
        sfeat[(size_t)p * CTOT + c] = (_Float16)v;
    }
    __syncthreads();

    const int wave = tid >> 5, lane = tid & 31;
    const int lrow = lane & 15, lhi = lane >> 4;

#pragma unroll
    for (int mi = 0; mi < MPW; ++mi) {
        const int mt = wave + mi * 8;

        v16h afrag[KB];
#pragma unroll
        for (int kb = 0; kb < KB; ++kb) {
            const _Float16* wrow =
                Wh + (size_t)(mt * 16 + lrow) * CTOT + kb * 32 + lhi * 8;
#pragma unroll
            for (int e = 0; e < 16; ++e)
                afrag[kb][e] = wrow[(e & 7) + ((e >> 3) << 4)];
        }

        const int mbase = mt * 16 + lhi * 8;
        float rsum[8], rsq[8];
#pragma unroll
        for (int r = 0; r < 8; ++r) { rsum[r] = 0.f; rsq[r] = 0.f; }

#pragma unroll
        for (int pt = 0; pt < PT; ++pt) {
            v8f acc = {};
#pragma unroll
            for (int kb = 0; kb < KB; ++kb) {
                v16h bf;
                const _Float16* fp =
                    sfeat + (size_t)(pt * 16 + lrow) * CTOT + kb * 32 + lhi * 8;
#pragma unroll
                for (int e = 0; e < 16; ++e) bf[e] = fp[(e & 7) + ((e >> 3) << 4)];
                acc = __builtin_amdgcn_wmma_f32_16x16x32_f16(
                    false, afrag[kb], false, bf, (short)0, acc, false, false);
            }
            const int n = n0 + pt * 16 + lrow;
#pragma unroll
            for (int r = 0; r < 8; ++r) {
                const float y = acc[r];
                ybuf[((size_t)b * COUT + (mbase + r)) * Nn + n] = y;
                rsum[r] += y;
                rsq[r]  += y * y;
            }
        }
#pragma unroll
        for (int r = 0; r < 8; ++r) {
            atomicAdd(&ssum[mbase + r], rsum[r]);
            atomicAdd(&ssq[mbase + r],  rsq[r]);
        }
    }
    __syncthreads();
    for (int ch = tid; ch < COUT; ch += 256) {
        atomicAdd(&gsum[ch],   ssum[ch]);
        atomicAdd(&gsumsq[ch], ssq[ch]);
    }
}

// ---------------------------------------------------------------------------
// Small helpers
// ---------------------------------------------------------------------------
__global__ void zero_f32(float* __restrict__ p, int n) {
    int i = blockIdx.x * blockDim.x + threadIdx.x;
    if (i < n) p[i] = 0.f;
}

__global__ void cvt_w(const float* __restrict__ W, _Float16* __restrict__ Wh,
                      int Cout, int Kin, int Kpad) {
    int i = blockIdx.x * blockDim.x + threadIdx.x;
    if (i < Cout * Kpad) {
        int r = i / Kpad, c = i % Kpad;
        Wh[i] = (_Float16)(c < Kin ? W[(size_t)r * Kin + c] : 0.f);
    }
}

__global__ void finalize_stats(const float* __restrict__ gsum,
                               const float* __restrict__ gsumsq,
                               const float* __restrict__ gamma,
                               const float* __restrict__ beta,
                               float* __restrict__ scale,
                               float* __restrict__ shift,
                               int C, float invP) {
    int o = blockIdx.x * blockDim.x + threadIdx.x;
    if (o < C) {
        float m = gsum[o] * invP;
        float v = gsumsq[o] * invP - m * m;
        float s = gamma[o] * rsqrtf(v + EPS_BN);
        scale[o] = s;
        shift[o] = beta[o] - m * s;
    }
}

__global__ void apply_bn_act(const float* __restrict__ y,
                             const float* __restrict__ scale,
                             const float* __restrict__ shift,
                             float* __restrict__ out,
                             int Nn, int C, long long total) {
    long long i = (long long)blockIdx.x * blockDim.x + threadIdx.x;
    if (i < total) {
        int ch = (int)((i / Nn) % C);
        float v = scale[ch] * y[i] + shift[ch];
        out[i] = v >= 0.f ? v : SLOPE * v;
    }
}

// ---------------------------------------------------------------------------
extern "C" void kernel_launch(void* const* d_in, const int* in_sizes, int n_in,
                              void* d_out, int out_size, void* d_ws, size_t ws_size,
                              hipStream_t stream) {
    (void)in_sizes; (void)n_in; (void)out_size; (void)ws_size;
    const float*     x   = (const float*)d_in[0];
    const long long* idx = (const long long*)d_in[1];
    const float* W1 = (const float*)d_in[2];
    const float* g1 = (const float*)d_in[3];
    const float* b1 = (const float*)d_in[4];
    const float* W2 = (const float*)d_in[5];
    const float* g2 = (const float*)d_in[6];
    const float* b2 = (const float*)d_in[7];
    const float* W3 = (const float*)d_in[8];
    const float* g3 = (const float*)d_in[9];
    const float* b3 = (const float*)d_in[10];
    const float* W4 = (const float*)d_in[11];
    const float* g4 = (const float*)d_in[12];
    const float* b4 = (const float*)d_in[13];

    constexpr int B = 8, N = 2048;
    // ---- workspace layout (all L2-resident: ~32.2 MiB total) ----
    float* x1   = (float*)d_ws;                       // B*64*N
    float* x2   = x1 + (size_t)B * 64 * N;            // B*64*N
    float* x3   = x2 + (size_t)B * 64 * N;            // B*128*N
    float* ybuf = x3 + (size_t)B * 128 * N;           // B*256*N (ymax / full y)
    float* stats = ybuf + (size_t)B * 256 * N;        // 1024 floats
    float* gsum   = stats;
    float* gsq    = stats + 256;
    float* scale  = stats + 512;
    float* shiftv = stats + 768;
    _Float16* W1h = (_Float16*)(stats + 1024);        // 64*32
    _Float16* W2h = W1h + 64 * 32;                    // 64*128
    _Float16* W3h = W2h + 64 * 128;                   // 128*128
    _Float16* W4h = W3h + 128 * 128;                  // 256*256

    // ---- convert / pad weights to f16 once per call ----
    cvt_w<<<(64 * 32 + 255) / 256, 256, 0, stream>>>(W1, W1h, 64, 6, 32);
    cvt_w<<<(64 * 128 + 255) / 256, 256, 0, stream>>>(W2, W2h, 64, 128, 128);
    cvt_w<<<(128 * 128 + 255) / 256, 256, 0, stream>>>(W3, W3h, 128, 128, 128);
    cvt_w<<<(256 * 256 + 255) / 256, 256, 0, stream>>>(W4, W4h, 256, 256, 256);

    const float invP2d = 1.f / (8.f * 2048.f * 20.f);
    const float invP1d = 1.f / (8.f * 2048.f);
    const int ecBlocks = B * (N / 16);   // 1024

    // ---- layer 1: 6 -> 64 ----
    zero_f32<<<2, 256, 0, stream>>>(gsum, 512);
    {
        size_t sh = (size_t)320 * 32 * sizeof(_Float16)
                  + 64 * 16 * 4 + 2 * 64 * 4 + 16 * 3 * 4;
        edgeconv_wmma<3, 64, 32><<<ecBlocks, 256, sh, stream>>>(
            x, idx, W1h, ybuf, gsum, gsq, N);
    }
    finalize_stats<<<1, 256, 0, stream>>>(gsum, gsq, g1, b1, scale, shiftv, 64, invP2d);
    {
        long long t = (long long)B * 64 * N;
        apply_bn_act<<<(unsigned)((t + 255) / 256), 256, 0, stream>>>(
            ybuf, scale, shiftv, x1, N, 64, t);
    }

    // ---- layer 2: 128 -> 64 ----
    zero_f32<<<2, 256, 0, stream>>>(gsum, 512);
    {
        size_t sh = (size_t)320 * 128 * sizeof(_Float16)
                  + 64 * 16 * 4 + 2 * 64 * 4 + 16 * 64 * 4;
        edgeconv_wmma<64, 64, 128><<<ecBlocks, 256, sh, stream>>>(
            x1, idx, W2h, ybuf, gsum, gsq, N);
    }
    finalize_stats<<<1, 256, 0, stream>>>(gsum, gsq, g2, b2, scale, shiftv, 64, invP2d);
    {
        long long t = (long long)B * 64 * N;
        apply_bn_act<<<(unsigned)((t + 255) / 256), 256, 0, stream>>>(
            ybuf, scale, shiftv, x2, N, 64, t);
    }

    // ---- layer 3: 128 -> 128 ----
    zero_f32<<<2, 256, 0, stream>>>(gsum, 512);
    {
        size_t sh = (size_t)320 * 128 * sizeof(_Float16)
                  + 128 * 16 * 4 + 2 * 128 * 4 + 16 * 64 * 4;
        edgeconv_wmma<64, 128, 128><<<ecBlocks, 256, sh, stream>>>(
            x2, idx, W3h, ybuf, gsum, gsq, N);
    }
    finalize_stats<<<1, 256, 0, stream>>>(gsum, gsq, g3, b3, scale, shiftv, 128, invP2d);
    {
        long long t = (long long)B * 128 * N;
        apply_bn_act<<<(unsigned)((t + 255) / 256), 256, 0, stream>>>(
            ybuf, scale, shiftv, x3, N, 128, t);
    }

    // ---- layer 4: concat(256) -> 256 (1d) ----
    zero_f32<<<2, 256, 0, stream>>>(gsum, 512);
    {
        size_t sh = (size_t)64 * 256 * sizeof(_Float16) + 2 * 256 * 4;
        conv1d_wmma<256, 256><<<B * (N / 64), 256, sh, stream>>>(
            x1, x2, x3, W4h, ybuf, gsum, gsq, N);
    }
    finalize_stats<<<1, 256, 0, stream>>>(gsum, gsq, g4, b4, scale, shiftv, 256, invP1d);
    {
        long long t = (long long)B * 256 * N;
        apply_bn_act<<<(unsigned)((t + 255) / 256), 256, 0, stream>>>(
            ybuf, scale, shiftv, (float*)d_out, N, 256, t);
    }
}
